// MoEDispatcher_19731079758695
// MI455X (gfx1250) — matrix operand order repaired and checked by reading, not compiled, not run on hardware
//
#include <hip/hip_runtime.h>
#include <hip/hip_bf16.h>

#define NN 8192
#define EE 8
#define KK 2
#define DD 1024
#define HH 1024

#define TM 128
#define TN 128
#define TK 32
#define ASTRIDE 40   // bf16 elements per LDS row (32 + pad; conflict-free, 16B-aligned)

typedef __attribute__((ext_vector_type(16))) __bf16 v16bf;
typedef __attribute__((ext_vector_type(8)))  __bf16 v8bf;
typedef __attribute__((ext_vector_type(8)))  float  v8f;

union frag16 { v16bf v; v8bf h[2]; };

__device__ inline __bf16 f2bf(float f) {
    unsigned u = __builtin_bit_cast(unsigned, f);
    unsigned r = u + 0x7FFFu + ((u >> 16) & 1u);   // round-to-nearest-even
    unsigned short hs = (unsigned short)(r >> 16);
    return __builtin_bit_cast(__bf16, hs);
}

// ---------------------------------------------------------------------------
// One-shot f32 -> bf16 conversion (bandwidth-bound; bf16 copies live in L2).
// ---------------------------------------------------------------------------
__global__ __launch_bounds__(256) void cvt_bf16(
    const float* __restrict__ src, __bf16* __restrict__ dst)
{
    long i = ((long)blockIdx.x * blockDim.x + threadIdx.x) * 8;
    float4 a = *(const float4*)(src + i);
    float4 b = *(const float4*)(src + i + 4);
    v8bf v = {};
    v[0]=f2bf(a.x); v[1]=f2bf(a.y); v[2]=f2bf(a.z); v[3]=f2bf(a.w);
    v[4]=f2bf(b.x); v[5]=f2bf(b.y); v[6]=f2bf(b.z); v[7]=f2bf(b.w);
    *(v8bf*)(dst + i) = v;
}

// ---------------------------------------------------------------------------
// Gating: top-3 of 8, softmax over top-2, dense gates, erf load reduction,
// and compaction of (token, gate) into per-expert row lists.
// ---------------------------------------------------------------------------
__global__ __launch_bounds__(256) void moe_gate(
    const float* __restrict__ logits, const float* __restrict__ clean,
    const float* __restrict__ nstd, float* __restrict__ gates,
    float* __restrict__ load, int* __restrict__ counts,
    int* __restrict__ row_idx, float* __restrict__ row_gate)
{
    __shared__ float lload[EE];
    const int tid = threadIdx.x;
    if (tid < EE) lload[tid] = 0.0f;
    __syncthreads();

    const int n = blockIdx.x * blockDim.x + tid;

    float l[EE];
    #pragma unroll
    for (int e = 0; e < EE; ++e) l[e] = logits[n * EE + e];

    int i0 = 0; float v0 = l[0];
    #pragma unroll
    for (int e = 1; e < EE; ++e) if (l[e] > v0) { v0 = l[e]; i0 = e; }
    int i1 = -1; float v1 = -3.4e38f;
    #pragma unroll
    for (int e = 0; e < EE; ++e) if (e != i0 && l[e] > v1) { v1 = l[e]; i1 = e; }
    float v2 = -3.4e38f;
    #pragma unroll
    for (int e = 0; e < EE; ++e) if (e != i0 && e != i1 && l[e] > v2) v2 = l[e];

    float e1 = __expf(v1 - v0);
    float inv = 1.0f / (1.0f + e1);
    float g0 = inv;
    float g1 = e1 * inv;

    #pragma unroll
    for (int e = 0; e < EE; ++e) {
        float gv = (e == i0) ? g0 : ((e == i1) ? g1 : 0.0f);
        gates[n * EE + e] = gv;
    }

    const float in_thr = v2, out_thr = v1;
    #pragma unroll
    for (int e = 0; e < EE; ++e) {
        float thr = (l[e] > in_thr) ? in_thr : out_thr;
        float x = (clean[n * EE + e] - thr) / nstd[n * EE + e];
        float p = 0.5f * (1.0f + erff(x * 0.70710678118654752f));
        atomicAdd(&lload[e], p);
    }

    int p0 = atomicAdd(&counts[i0], 1);
    row_idx[i0 * NN + p0] = n;  row_gate[i0 * NN + p0] = g0;
    int p1 = atomicAdd(&counts[i1], 1);
    row_idx[i1 * NN + p1] = n;  row_gate[i1 * NN + p1] = g1;

    __syncthreads();
    if (tid < EE) atomicAdd(&load[tid], lload[tid]);
}

// ---------------------------------------------------------------------------
// Grouped GEMM on pre-converted bf16: out[tok,:] += gate * net[tok,:] @ W[e].
// WMMA 16x16x32 bf16 -> f32. Block 128x128, 8 waves (2x4), wave = 64x32.
// Double-buffered LDS, one barrier per K-step, global prefetch in registers.
// ---------------------------------------------------------------------------
__global__ __launch_bounds__(256) void moe_gemm(
    const __bf16* __restrict__ netbf, const __bf16* __restrict__ wbf,
    const int* __restrict__ counts, const int* __restrict__ row_idx,
    const float* __restrict__ row_gate, float* __restrict__ out)
{
    const int e  = blockIdx.x >> 6;            // NN/TM = 64 tiles per expert
    const int tm = (blockIdx.x & 63) * TM;
    const int cnt = counts[e];
    if (tm >= cnt) return;                      // uniform early exit
    const int n0 = blockIdx.y * TN;

    __shared__ alignas(16) __bf16 As[2][TM * ASTRIDE];   // row-major, padded
    __shared__ alignas(16) __bf16 Bs[2][TN * ASTRIDE];   // column-major
    __shared__ int   sTok[TM];
    __shared__ float sGate[TM];

    const int tid  = threadIdx.x;
    const int lane = tid & 31;
    const int w    = tid >> 5;
    const int wm   = w >> 2;        // 0..1  -> 64-row slab
    const int wn   = w & 3;         // 0..3  -> 32-col slab
    const int lrow = lane & 15;

    if (tid < TM) {
        int gm = tm + tid;
        if (gm < cnt) {
            sTok[tid]  = row_idx[e * NN + gm];
            sGate[tid] = row_gate[e * NN + gm];
        } else { sTok[tid] = -1; sGate[tid] = 0.0f; }
    }
    __syncthreads();

    // A staging: rows (tid>>2) and 64+(tid>>2), K-chunk (tid&3)*8
    const int  arow0 = tid >> 2;
    const int  arow1 = 64 + arow0;
    const int  akk   = (tid & 3) * 8;
    const long atok0 = sTok[arow0];
    const long atok1 = sTok[arow1];

    // B staging: k = tid>>3, 16 columns starting at (tid&7)*16
    const int bk  = tid >> 3;
    const int bc0 = (tid & 7) * 16;
    const __bf16* wbase = wbf + ((long)e * DD + bk) * HH + n0 + bc0;

    const v8bf z8 = {};
    v8bf aR0, aR1, bR0, bR1;

    #define LOAD_STAGE(kc)                                                     \
        do {                                                                   \
            aR0 = (atok0 >= 0) ? *(const v8bf*)(netbf + atok0 * DD + (kc) + akk) : z8; \
            aR1 = (atok1 >= 0) ? *(const v8bf*)(netbf + atok1 * DD + (kc) + akk) : z8; \
            const __bf16* wp = wbase + (long)(kc) * HH;                        \
            bR0 = *(const v8bf*)wp;                                            \
            bR1 = *(const v8bf*)(wp + 8);                                      \
        } while (0)

    #define STORE_STAGE(p)                                                     \
        do {                                                                   \
            *(v8bf*)&As[p][arow0 * ASTRIDE + akk] = aR0;                       \
            *(v8bf*)&As[p][arow1 * ASTRIDE + akk] = aR1;                       \
            _Pragma("unroll")                                                  \
            for (int j = 0; j < 8; ++j) Bs[p][(bc0 + j) * ASTRIDE + bk] = bR0[j]; \
            _Pragma("unroll")                                                  \
            for (int j = 0; j < 8; ++j) Bs[p][(bc0 + 8 + j) * ASTRIDE + bk] = bR1[j]; \
        } while (0)

    #define COMPUTE(p)                                                         \
        do {                                                                   \
            frag16 a[4], b[2];                                                 \
            _Pragma("unroll")                                                  \
            for (int i = 0; i < 4; ++i) {                                      \
                int row = wm * 64 + i * 16 + lrow;                             \
                a[i].h[0] = *(const v8bf*)&As[p][row * ASTRIDE + aks];         \
                a[i].h[1] = *(const v8bf*)&As[p][row * ASTRIDE + aks + 16];    \
            }                                                                  \
            _Pragma("unroll")                                                  \
            for (int j = 0; j < 2; ++j) {                                      \
                int col = wn * 32 + j * 16 + lrow;                             \
                b[j].h[0] = *(const v8bf*)&Bs[p][col * ASTRIDE + bks];         \
                b[j].h[1] = *(const v8bf*)&Bs[p][col * ASTRIDE + bks + 16];    \
            }                                                                  \
            _Pragma("unroll")                                                  \
            for (int i = 0; i < 4; ++i)                                        \
                _Pragma("unroll")                                              \
                for (int j = 0; j < 2; ++j)                                    \
                    acc[i][j] = __builtin_amdgcn_wmma_f32_16x16x32_bf16(       \
                        false, a[i].v, false, b[j].v, (short)0, acc[i][j],     \
                        false, false);                                         \
        } while (0)

    const v8f vzero = {};
    v8f acc[4][2];
    #pragma unroll
    for (int i = 0; i < 4; ++i)
        #pragma unroll
        for (int j = 0; j < 2; ++j) acc[i][j] = vzero;

    // fragment half-selects per the documented 16-bit A/B VGPR layouts
    const int aks = (lane < 16) ? 0 : 8;     // A: lanes 0-15 -> K 0-7/16-23
    const int bks = (lane < 16) ? 0 : 8;     // B: lanes 0-15 -> K 0-7/16-23

    // prologue: fill buffer 0
    LOAD_STAGE(0);
    STORE_STAGE(0);

    #pragma unroll 1
    for (int kb = 0; kb < DD; kb += 2 * TK) {
        // ---- stage A: compute buffer 0, prefetch+fill buffer 1 ----
        __syncthreads();
        LOAD_STAGE(kb + TK);          // kb+TK < DD always holds here
        COMPUTE(0);
        STORE_STAGE(1);
        // ---- stage B: compute buffer 1, prefetch+fill buffer 0 ----
        __syncthreads();
        if (kb + 2 * TK < DD) LOAD_STAGE(kb + 2 * TK);
        COMPUTE(1);
        if (kb + 2 * TK < DD) STORE_STAGE(0);
    }
    #undef LOAD_STAGE
    #undef STORE_STAGE
    #undef COMPUTE

    // ---- epilogue: gate-scale and scatter-accumulate ----
    #pragma unroll
    for (int i = 0; i < 4; ++i) {
        int rbase = wm * 64 + i * 16 + ((lane >> 4) << 3);
        #pragma unroll
        for (int j = 0; j < 2; ++j) {
            int col = n0 + wn * 32 + j * 16 + lrow;
            #pragma unroll
            for (int r = 0; r < 8; ++r) {
                int m = rbase + r;
                int tok = sTok[m];
                if (tok >= 0)
                    atomicAdd(out + (long)tok * HH + col, sGate[m] * acc[i][j][r]);
            }
        }
    }
}

// ---------------------------------------------------------------------------
extern "C" void kernel_launch(void* const* d_in, const int* in_sizes, int n_in,
                              void* d_out, int out_size, void* d_ws, size_t ws_size,
                              hipStream_t stream) {
    (void)in_sizes; (void)n_in; (void)out_size; (void)ws_size;
    const float* net    = (const float*)d_in[0];
    const float* logits = (const float*)d_in[1];
    const float* clean  = (const float*)d_in[2];
    const float* nstd   = (const float*)d_in[3];
    const float* W      = (const float*)d_in[4];

    float* out   = (float*)d_out;
    float* gates = out + (size_t)NN * HH;
    float* load  = gates + (size_t)NN * EE;

    char*  ws       = (char*)d_ws;
    int*   counts   = (int*)ws;                                   // EE ints
    int*   row_idx  = (int*)(ws + 256);                           // EE*NN ints
    float* row_gate = (float*)(ws + 256 + sizeof(int) * EE * NN); // EE*NN floats
    __bf16* netbf   = (__bf16*)(ws + (1u << 20));                 // 16 MB
    __bf16* wbf     = (__bf16*)(ws + (1u << 20) + sizeof(__bf16) * (size_t)NN * DD); // 16 MB

    hipMemsetAsync(out, 0, sizeof(float) * (size_t)NN * HH, stream);
    hipMemsetAsync(load, 0, sizeof(float) * EE, stream);
    hipMemsetAsync(counts, 0, sizeof(int) * EE, stream);

    // bf16 pre-conversion (both arrays are 8.39M elements -> 4096 blocks each)
    cvt_bf16<<<(NN * DD) / (256 * 8), 256, 0, stream>>>(net, netbf);
    cvt_bf16<<<(EE * DD * HH) / (256 * 8), 256, 0, stream>>>(W, wbf);

    moe_gate<<<NN / 256, 256, 0, stream>>>(logits, clean, nstd, gates, load,
                                           counts, row_idx, row_gate);

    dim3 grid(EE * (NN / TM), HH / TN);
    moe_gemm<<<grid, 256, 0, stream>>>(netbf, wbf, counts, row_idx, row_gate, out);
}
